// FlowAwareGraphConv_90537910599955
// MI455X (gfx1250) — matrix operand order repaired and checked by reading, not compiled, not run on hardware
//
#include <hip/hip_runtime.h>

typedef __attribute__((ext_vector_type(16))) _Float16 v16h;
typedef __attribute__((ext_vector_type(2)))  _Float16 v2h;
typedef __attribute__((ext_vector_type(8)))  float    v8f;

#define DIM 64

// ---------------------------------------------------------------------------
// Kernel 1: dual GEMM  h_self = x@W_self + b_self  (-> d_out)
//                      h_neigh = x@W_neigh + b_neigh (-> ws)
// One wave computes a 16-row x 64-col tile for BOTH weight matrices,
// reusing the A (x) fragments. Weights are staged in LDS pre-swizzled into
// WMMA B-fragment layout so the hot loop does one contiguous 32B LDS read
// per fragment per lane.
// ---------------------------------------------------------------------------
__global__ __launch_bounds__(256) void gemm_dual_kernel(
    const float* __restrict__ x,
    const float* __restrict__ Wself,  const float* __restrict__ bself,
    const float* __restrict__ Wneigh, const float* __restrict__ bneigh,
    float* __restrict__ hself,   // d_out
    float* __restrict__ hneigh,  // scratch
    int N)
{
    // Pre-swizzled B fragments: frag index f = ((mat*4 + nt)*2 + t),
    // per-lane v16h, element e -> W[mat][kb+e][n] with
    //   kb = 32*t + 16*(lane>>4),  n = 16*nt + (lane&15)
    __shared__ v16h Wfrag[16][32];   // 16 KB

    const int tid = threadIdx.x;
    {
        v2h* Wp = (v2h*)Wfrag;       // 4096 packed half2 elements
        for (int i = tid; i < 16 * 32 * 8; i += 256) {
            const int f    = i >> 8;          // fragment id
            const int lane = (i >> 3) & 31;
            const int e    = (i & 7) * 2;     // even element of the pair
            const int mat  = f >> 3;
            const int nt   = (f >> 1) & 3;
            const int t    = f & 1;
            const int kb   = t * 32 + ((lane >> 4) << 4);
            const int n    = nt * 16 + (lane & 15);
            const float* Wsrc = mat ? Wneigh : Wself;
            v2h p;
            p.x = (_Float16)Wsrc[(kb + e) * DIM + n];
            p.y = (_Float16)Wsrc[(kb + e + 1) * DIM + n];
            Wp[i] = p;
        }
    }
    __syncthreads();

    const int lane = tid & 31;
    const int wave = tid >> 5;
    const int row0 = (blockIdx.x * 8 + wave) * 16;
    if (row0 >= N) return;

    const int m  = lane & 15;          // row within tile / column within tile
    const int hi = (lane >> 4) & 1;    // half-wave select

    // --- A fragments for k-tiles 0 and 1 (contiguous k order, same as B staging) -------
    // lanes 0-15: K = {0..15}+32t ; lanes 16-31: K = {16..31}+32t
    v16h afrag[2];
    int r = row0 + m; if (r > N - 1) r = N - 1;   // clamp tail loads
#pragma unroll
    for (int t = 0; t < 2; ++t) {
        const float* xr = x + (size_t)r * DIM + t * 32 + hi * 16;
        float4 c0 = *(const float4*)(xr);
        float4 c1 = *(const float4*)(xr + 4);
        float4 c2 = *(const float4*)(xr + 8);
        float4 c3 = *(const float4*)(xr + 12);
        afrag[t][0]  = (_Float16)c0.x; afrag[t][1]  = (_Float16)c0.y;
        afrag[t][2]  = (_Float16)c0.z; afrag[t][3]  = (_Float16)c0.w;
        afrag[t][4]  = (_Float16)c1.x; afrag[t][5]  = (_Float16)c1.y;
        afrag[t][6]  = (_Float16)c1.z; afrag[t][7]  = (_Float16)c1.w;
        afrag[t][8]  = (_Float16)c2.x; afrag[t][9]  = (_Float16)c2.y;
        afrag[t][10] = (_Float16)c2.z; afrag[t][11] = (_Float16)c2.w;
        afrag[t][12] = (_Float16)c3.x; afrag[t][13] = (_Float16)c3.y;
        afrag[t][14] = (_Float16)c3.z; afrag[t][15] = (_Float16)c3.w;
    }

    const bool fullTile = (row0 + 16 <= N);   // wave-uniform fast path

#pragma unroll
    for (int mat = 0; mat < 2; ++mat) {
        const float* bias = mat ? bneigh : bself;
        float* dst        = mat ? hneigh : hself;
#pragma unroll
        for (int nt = 0; nt < 4; ++nt) {
            const int nn = nt * 16 + m;
            v8f acc = {};
#pragma unroll
            for (int t = 0; t < 2; ++t) {
                const v16h b = Wfrag[((mat * 4 + nt) << 1) + t][lane];
                acc = __builtin_amdgcn_wmma_f32_16x16x32_f16(
                    false, afrag[t], false, b, (short)0, acc, false, false);
            }
            const float bv = bias[nn];
            // C/D layout: VGPR r8 -> M = row0 + r8 + 8*hi, N = nn
            float* dstp = dst + (size_t)(row0 + hi * 8) * DIM + nn;
            if (fullTile) {
#pragma unroll
                for (int r8 = 0; r8 < 8; ++r8)
                    dstp[r8 * DIM] = acc[r8] + bv;   // imm-offset stores
            } else {
#pragma unroll
                for (int r8 = 0; r8 < 8; ++r8)
                    if (row0 + hi * 8 + r8 < N)
                        dstp[r8 * DIM] = acc[r8] + bv;
            }
        }
    }
}

// ---------------------------------------------------------------------------
// Kernel 2: per-edge gated message + scatter-add.
// One wave per edge; each lane owns 2 features (float2) -> coalesced 256B
// row gather from h_neigh (L2-resident: 25.6MB << 192MB L2) and 64 f32
// atomics into the destination row of d_out (also L2-resident).
// ---------------------------------------------------------------------------
__global__ __launch_bounds__(256) void edge_kernel(
    const long long* __restrict__ eidx,   // [2, E] int64: row=dst, col=src
    const float* __restrict__ ew,
    const float* __restrict__ Wedge,      // [64]
    const float* __restrict__ hneigh,
    float* __restrict__ out,
    int E)
{
    const int lane = threadIdx.x & 31;
    const int e    = blockIdx.x * 8 + (threadIdx.x >> 5);
    if (e >= E) return;

    const long long dstRow = eidx[e];
    const long long srcCol = eidx[(size_t)E + e];
    const float     w      = ew[e];

    const float2 we = ((const float2*)Wedge)[lane];
    const float2 hn = *(const float2*)(hneigh + (size_t)srcCol * DIM + lane * 2);

    const float g0 = 1.0f / (1.0f + __expf(-w * we.x));
    const float g1 = 1.0f / (1.0f + __expf(-w * we.y));

    float* o = out + (size_t)dstRow * DIM + lane * 2;
    atomicAdd(o,     hn.x * g0);
    atomicAdd(o + 1, hn.y * g1);
}

// ---------------------------------------------------------------------------
// Kernel 3: in-place LayerNorm + LeakyReLU(0.2). One wave per row; mean/var
// via 5-step shfl_xor wave32 reduction.
// ---------------------------------------------------------------------------
__global__ __launch_bounds__(256) void ln_kernel(
    float* __restrict__ out,
    const float* __restrict__ gamma,
    const float* __restrict__ beta,
    int N)
{
    const int lane = threadIdx.x & 31;
    const int row  = blockIdx.x * 8 + (threadIdx.x >> 5);
    if (row >= N) return;

    float2 v = *(const float2*)(out + (size_t)row * DIM + lane * 2);
    float s  = v.x + v.y;
    float sq = v.x * v.x + v.y * v.y;
#pragma unroll
    for (int off = 16; off >= 1; off >>= 1) {
        s  += __shfl_xor(s,  off, 32);
        sq += __shfl_xor(sq, off, 32);
    }
    const float mu   = s * (1.0f / 64.0f);
    const float var  = sq * (1.0f / 64.0f) - mu * mu;
    const float rstd = rsqrtf(var + 1e-5f);

    const float2 g = ((const float2*)gamma)[lane];
    const float2 b = ((const float2*)beta)[lane];
    float y0 = (v.x - mu) * rstd * g.x + b.x;
    float y1 = (v.y - mu) * rstd * g.y + b.y;
    y0 = (y0 >= 0.0f) ? y0 : 0.2f * y0;
    y1 = (y1 >= 0.0f) ? y1 : 0.2f * y1;
    *(float2*)(out + (size_t)row * DIM + lane * 2) = make_float2(y0, y1);
}

// ---------------------------------------------------------------------------
extern "C" void kernel_launch(void* const* d_in, const int* in_sizes, int n_in,
                              void* d_out, int out_size, void* d_ws, size_t ws_size,
                              hipStream_t stream)
{
    const float*     x      = (const float*)d_in[0];
    const long long* eidx   = (const long long*)d_in[1];   // int64 per reference
    const float*     ew     = (const float*)d_in[2];
    const float*     Wself  = (const float*)d_in[3];
    const float*     bself  = (const float*)d_in[4];
    const float*     Wneigh = (const float*)d_in[5];
    const float*     bneigh = (const float*)d_in[6];
    const float*     Wedge  = (const float*)d_in[7];
    const float*     gamma  = (const float*)d_in[8];
    const float*     beta   = (const float*)d_in[9];

    float* out    = (float*)d_out;
    float* hneigh = (float*)d_ws;        // N*64 f32 = 25.6 MB scratch

    const int N = in_sizes[0] / DIM;
    const int E = in_sizes[2];

    // 1) Dual WMMA GEMM: overwrites d_out with h_self (acts as agg init too)
    const int gemmBlocks = (N + 127) / 128;   // 8 waves/block * 16 rows/wave
    gemm_dual_kernel<<<gemmBlocks, 256, 0, stream>>>(
        x, Wself, bself, Wneigh, bneigh, out, hneigh, N);

    // 2) Edge gather + gate + scatter-add into d_out
    const int edgeBlocks = (E + 7) / 8;       // 1 wave per edge
    edge_kernel<<<edgeBlocks, 256, 0, stream>>>(eidx, ew, Wedge, hneigh, out, E);

    // 3) In-place LayerNorm + LeakyReLU
    const int lnBlocks = (N + 7) / 8;         // 1 wave per row
    ln_kernel<<<lnBlocks, 256, 0, stream>>>(out, gamma, beta, N);
}
